// MemoryEfficientCrossAttention_56684978372660
// MI455X (gfx1250) — matrix operand order repaired and checked by reading, not compiled, not run on hardware
//
#include <hip/hip_runtime.h>
#include <cstdint>
#include <cstddef>

typedef __attribute__((ext_vector_type(16))) _Float16 v16h;
typedef __attribute__((ext_vector_type(8)))  _Float16 v8h;
typedef __attribute__((ext_vector_type(8)))  float    v8f;

#define CAT16(lo, hi) __builtin_shufflevector((lo), (hi), 0,1,2,3,4,5,6,7,8,9,10,11,12,13,14,15)
#define WMMA_F16(a, b, c) \
  __builtin_amdgcn_wmma_f32_16x16x32_f16(false, (a), false, (b), (short)0, (c), false, false)

namespace {
constexpr int kB  = 2;
constexpr int kS  = 4096;
constexpr int kQD = 320;
constexpr int kH  = 8;
constexpr int kD  = 64;
constexpr int kIn = 512;           // H * D
constexpr int kM  = kB * kS;       // 8192 token rows
constexpr int kChunks = kS / 32;   // 32-key chunks per head
}

// ---- CDNA5 async global->LDS copy (ASYNCcnt-tracked, no VGPR data path) ----
typedef __attribute__((address_space(3))) uint8_t lds_byte_t;

__device__ __forceinline__ uint32_t lds_off(const void* p) {
  // addrspace(3) pointer value == byte offset from wave LDS base, which is
  // exactly what VDST of global_load_async_to_lds expects.
  return (uint32_t)(uintptr_t)(lds_byte_t*)p;
}

__device__ __forceinline__ void async_copy_b128(uint32_t dst_lds, const void* src) {
  asm volatile("global_load_async_to_lds_b128 %0, %1, off"
               :
               : "v"(dst_lds), "v"((uint64_t)(uintptr_t)src)
               : "memory");
}

__device__ __forceinline__ void wait_async0() {
  asm volatile("s_wait_asynccnt 0" ::: "memory");
}

__device__ __forceinline__ float redmax16(float v) {
  v = fmaxf(v, __shfl_xor(v, 1, 32));
  v = fmaxf(v, __shfl_xor(v, 2, 32));
  v = fmaxf(v, __shfl_xor(v, 4, 32));
  v = fmaxf(v, __shfl_xor(v, 8, 32));
  return v;
}
__device__ __forceinline__ float redsum16(float v) {
  v += __shfl_xor(v, 1, 32);
  v += __shfl_xor(v, 2, 32);
  v += __shfl_xor(v, 4, 32);
  v += __shfl_xor(v, 8, 32);
  return v;
}

__global__ void cvt_f32_f16(const float* __restrict__ in, _Float16* __restrict__ out, int n) {
  int i = blockIdx.x * blockDim.x + threadIdx.x;
  int stride = gridDim.x * blockDim.x;
  for (; i < n; i += stride) out[i] = (_Float16)in[i];
}

// ---------------------------------------------------------------------------
// QKV projection: [8192,320]f16 x [320,512]f16 -> per-head f16 layouts.
// blockIdx.z selects Wq/Wk/Wv. Q gets 1/sqrt(64)=0.125 folded in.
// Q,K: [b,h,s,64]; V stored transposed: [b,h,64,s].
// ---------------------------------------------------------------------------
__global__ void __launch_bounds__(128)
qkv_gemm(const _Float16* __restrict__ xh,
         const _Float16* __restrict__ wqh,
         const _Float16* __restrict__ wkh,
         const _Float16* __restrict__ wvh,
         _Float16* __restrict__ qo,
         _Float16* __restrict__ ko,
         _Float16* __restrict__ vto) {
  const int which = blockIdx.z;
  const _Float16* __restrict__ W = (which == 0) ? wqh : (which == 1) ? wkh : wvh;
  const int m0 = blockIdx.x * 64;
  const int n0 = blockIdx.y * 64;
  const int t = threadIdx.x;
  const int lane = t & 31, wid = t >> 5;
  const int lm = lane & 15, l16 = lane >> 4;

  __shared__ _Float16 At[64][32];     // [m_local][k_local], k contiguous
  __shared__ _Float16 BtT[64][32];    // [n_local][k_local], transposed at fill

  v8f acc[4] = {};

  for (int k0 = 0; k0 < kQD; k0 += 32) {
    __syncthreads();
    {  // A tile: 64x32 halves via async global->LDS (no transpose needed)
      const int row = t >> 1, cb = (t & 1) * 16;
      const _Float16* g = &xh[(size_t)(m0 + row) * kQD + k0 + cb];
      async_copy_b128(lds_off(&At[row][cb]), g);
      async_copy_b128(lds_off(&At[row][cb + 8]), g + 8);
    }
    {  // B tile: 32x64 halves loaded n-contiguous, stored transposed
      const int kr = t >> 2, nc = (t & 3) * 16;
      const v8h* g = (const v8h*)&W[(size_t)(k0 + kr) * kIn + n0 + nc];
      v8h b0 = g[0], b1 = g[1];
#pragma unroll
      for (int i = 0; i < 8; ++i) BtT[nc + i][kr]     = b0[i];
#pragma unroll
      for (int i = 0; i < 8; ++i) BtT[nc + 8 + i][kr] = b1[i];
    }
    wait_async0();
    __syncthreads();

    // A fragment: row = lm, halves 0..15 -> k=16*l16+0..15
    const int arow = wid * 16 + lm;
    v8h alo = *(const v8h*)&At[arow][16 * l16];
    v8h ahi = *(const v8h*)&At[arow][16 * l16 + 8];
    v16h af = CAT16(alo, ahi);
    v16h bf[4];
#pragma unroll
    for (int nt = 0; nt < 4; ++nt) {
      const int brow = nt * 16 + lm;  // B frag: n = lm (+16*nt), half h -> k = h + 16*l16
      v8h blo = *(const v8h*)&BtT[brow][16 * l16];
      v8h bhi = *(const v8h*)&BtT[brow][16 * l16 + 8];
      bf[nt] = CAT16(blo, bhi);
    }
#pragma unroll
    for (int nt = 0; nt < 4; ++nt) acc[nt] = WMMA_F16(af, bf[nt], acc[nt]);
  }

  const float scale = (which == 0) ? 0.125f : 1.0f;
#pragma unroll
  for (int nt = 0; nt < 4; ++nt) {
    const int n = n0 + nt * 16 + lm;
    const int hh = n >> 6, dd = n & 63;
#pragma unroll
    for (int r = 0; r < 8; ++r) {
      const int row = m0 + wid * 16 + r + 8 * l16;  // C layout: m = r + 8*(lane/16)
      const int bb = row >> 12, ss = row & (kS - 1);
      const _Float16 val = (_Float16)(acc[nt][r] * scale);
      if (which == 2) {
        vto[((size_t)(bb * kH + hh) * kD + dd) * kS + ss] = val;
      } else if (which == 0) {
        qo[((size_t)(bb * kH + hh) * kS + ss) * kD + dd] = val;
      } else {
        ko[((size_t)(bb * kH + hh) * kS + ss) * kD + dd] = val;
      }
    }
  }
}

// ---------------------------------------------------------------------------
// Flash attention: grid (S/64, B*H), 128 threads (4 waves, 16 q-rows each).
// Online softmax over 32-key chunks. K/V chunk staging is double-buffered
// through global_load_async_to_lds_b128 so the next chunk's HBM/L2 traffic
// overlaps the current chunk's WMMA + softmax work.
// ---------------------------------------------------------------------------
__global__ void __launch_bounds__(128)
flash_attn(const _Float16* __restrict__ Q,
           const _Float16* __restrict__ Kk,
           const _Float16* __restrict__ Vt,
           _Float16* __restrict__ attn) {
  const int bh = blockIdx.y;
  const int bb = bh >> 3, hh = bh & 7;
  const int q0 = blockIdx.x * 64;
  const int t = threadIdx.x;
  const int lane = t & 31, wid = t >> 5;
  const int lm = lane & 15, l16 = lane >> 4;

  __shared__ _Float16 Qt[64][64];         // q rows x head dim
  __shared__ _Float16 Kc[2][32][64];      // key chunk (d contiguous), double-buffered
  __shared__ _Float16 Vc[2][64][32];      // V chunk transposed [d][kk], double-buffered
  __shared__ _Float16 Pt[4][16][32];      // per-wave probabilities relayout

  const _Float16* __restrict__ Qb = Q + ((size_t)bh * kS + q0) * kD;
  const _Float16* __restrict__ Kb = Kk + (size_t)bh * kS * kD;
  const _Float16* __restrict__ Vb = Vt + (size_t)bh * kD * kS;

  // Per-thread fixed slots inside the cooperative tile fills.
  const int krow = t >> 2, kcb = (t & 3) * 16;   // K chunk: 32x64 halves
  const int vrow = t >> 1, vcb = (t & 1) * 16;   // V chunk: 64x32 halves

  {  // Q tile: 64x64 halves, async copy, 32 halves per thread
    const int row = t >> 1, cb = (t & 1) * 32;
    const _Float16* g = &Qb[(size_t)row * kD + cb];
    async_copy_b128(lds_off(&Qt[row][cb]),      g);
    async_copy_b128(lds_off(&Qt[row][cb + 8]),  g + 8);
    async_copy_b128(lds_off(&Qt[row][cb + 16]), g + 16);
    async_copy_b128(lds_off(&Qt[row][cb + 24]), g + 24);
  }
  {  // chunk 0 into buffer 0
    const _Float16* gk = &Kb[(size_t)krow * kD + kcb];
    async_copy_b128(lds_off(&Kc[0][krow][kcb]),     gk);
    async_copy_b128(lds_off(&Kc[0][krow][kcb + 8]), gk + 8);
    const _Float16* gv = &Vb[(size_t)vrow * kS + vcb];
    async_copy_b128(lds_off(&Vc[0][vrow][vcb]),     gv);
    async_copy_b128(lds_off(&Vc[0][vrow][vcb + 8]), gv + 8);
  }
  wait_async0();
  __syncthreads();

  // Persistent Q fragments for this wave (2 k-steps over D=64)
  const int arow = wid * 16 + lm;
  v16h aq[2];
#pragma unroll
  for (int ks = 0; ks < 2; ++ks) {
    v8h lo = *(const v8h*)&Qt[arow][32 * ks + 16 * l16];
    v8h hi = *(const v8h*)&Qt[arow][32 * ks + 16 * l16 + 8];
    aq[ks] = CAT16(lo, hi);
  }

  v8f O[4] = {};
  float rm[8], rs[8];
#pragma unroll
  for (int r = 0; r < 8; ++r) { rm[r] = -1e30f; rs[r] = 0.0f; }

  for (int i = 0; i < kChunks; ++i) {
    const int cur = i & 1;
    if (i) {           // chunk-i asyncs were issued last iteration
      wait_async0();   // our wave's copies for chunk i are in LDS
      __syncthreads(); // everyone's copies visible; spare buffer reads done
    }
    if (i + 1 < kChunks) {  // prefetch chunk i+1 into the spare buffer
      const int nxt = cur ^ 1;
      const int kk1 = (i + 1) * 32;
      const _Float16* gk = &Kb[(size_t)(kk1 + krow) * kD + kcb];
      async_copy_b128(lds_off(&Kc[nxt][krow][kcb]),     gk);
      async_copy_b128(lds_off(&Kc[nxt][krow][kcb + 8]), gk + 8);
      const _Float16* gv = &Vb[(size_t)vrow * kS + kk1 + vcb];
      async_copy_b128(lds_off(&Vc[nxt][vrow][vcb]),     gv);
      async_copy_b128(lds_off(&Vc[nxt][vrow][vcb + 8]), gv + 8);
    }

    // scores: 16 q-rows x 32 keys = 2 n-tiles x 2 k-steps
    v16h kf[2][2];
#pragma unroll
    for (int nt = 0; nt < 2; ++nt) {
      const int brow = nt * 16 + lm;  // B frag: n = key index, k = d
#pragma unroll
      for (int ks = 0; ks < 2; ++ks) {
        v8h lo = *(const v8h*)&Kc[cur][brow][32 * ks + 16 * l16];
        v8h hi = *(const v8h*)&Kc[cur][brow][32 * ks + 16 * l16 + 8];
        kf[nt][ks] = CAT16(lo, hi);
      }
    }
    v8f sc[2] = {};
#pragma unroll
    for (int ks = 0; ks < 2; ++ks) {
#pragma unroll
      for (int nt = 0; nt < 2; ++nt) sc[nt] = WMMA_F16(aq[ks], kf[nt][ks], sc[nt]);
    }

    // online softmax; each row lives across 16 lanes of a half-wave
#pragma unroll
    for (int r = 0; r < 8; ++r) {
      float s0 = sc[0][r], s1 = sc[1][r];
      float mnew = fmaxf(rm[r], redmax16(fmaxf(s0, s1)));
      float corr = __expf(rm[r] - mnew);
      rm[r] = mnew;
      float p0 = __expf(s0 - mnew);
      float p1 = __expf(s1 - mnew);
      rs[r] = rs[r] * corr + redsum16(p0 + p1);
#pragma unroll
      for (int dt = 0; dt < 4; ++dt) O[dt][r] *= corr;
      const int mrow = r + 8 * l16;
      Pt[wid][mrow][lm]      = (_Float16)p0;   // wave-private: DS in-order, no barrier
      Pt[wid][mrow][16 + lm] = (_Float16)p1;
    }

    // P fragment (A layout) from wave-private LDS tile
    v8h plo = *(const v8h*)&Pt[wid][lm][16 * l16];
    v8h phi = *(const v8h*)&Pt[wid][lm][16 * l16 + 8];
    v16h pf = CAT16(plo, phi);
    v16h vf[4];
#pragma unroll
    for (int dt = 0; dt < 4; ++dt) {
      const int brow = dt * 16 + lm;  // B frag from Vc: n = d, k = key (contiguous rows)
      v8h lo = *(const v8h*)&Vc[cur][brow][16 * l16];
      v8h hi = *(const v8h*)&Vc[cur][brow][16 * l16 + 8];
      vf[dt] = CAT16(lo, hi);
    }
#pragma unroll
    for (int dt = 0; dt < 4; ++dt) O[dt] = WMMA_F16(pf, vf[dt], O[dt]);
  }

#pragma unroll
  for (int r = 0; r < 8; ++r) {
    const float inv = 1.0f / rs[r];
    const int srow = q0 + wid * 16 + r + 8 * l16;
#pragma unroll
    for (int dt = 0; dt < 4; ++dt) {
      const int dd = dt * 16 + lm;
      attn[((size_t)(bb * kS + srow)) * kIn + hh * kD + dd] = (_Float16)(O[dt][r] * inv);
    }
  }
}

// ---------------------------------------------------------------------------
// Output projection: [8192,512]f16 x [512,320]f16 + bias -> f32 out
// ---------------------------------------------------------------------------
__global__ void __launch_bounds__(128)
out_gemm(const _Float16* __restrict__ Ah,
         const _Float16* __restrict__ Woh,
         const float* __restrict__ bo,
         float* __restrict__ out) {
  const int m0 = blockIdx.x * 64;
  const int n0 = blockIdx.y * 64;
  const int t = threadIdx.x;
  const int lane = t & 31, wid = t >> 5;
  const int lm = lane & 15, l16 = lane >> 4;

  __shared__ _Float16 At[64][32];
  __shared__ _Float16 BtT[64][32];

  v8f acc[4] = {};

  for (int k0 = 0; k0 < kIn; k0 += 32) {
    __syncthreads();
    {  // A tile via async copy
      const int row = t >> 1, cb = (t & 1) * 16;
      const _Float16* g = &Ah[(size_t)(m0 + row) * kIn + k0 + cb];
      async_copy_b128(lds_off(&At[row][cb]),     g);
      async_copy_b128(lds_off(&At[row][cb + 8]), g + 8);
    }
    {  // Wo tile, transposed at fill
      const int kr = t >> 2, nc = (t & 3) * 16;
      const v8h* g = (const v8h*)&Woh[(size_t)(k0 + kr) * kQD + n0 + nc];
      v8h b0 = g[0], b1 = g[1];
#pragma unroll
      for (int i = 0; i < 8; ++i) BtT[nc + i][kr]     = b0[i];
#pragma unroll
      for (int i = 0; i < 8; ++i) BtT[nc + 8 + i][kr] = b1[i];
    }
    wait_async0();
    __syncthreads();

    const int arow = wid * 16 + lm;
    v8h alo = *(const v8h*)&At[arow][16 * l16];
    v8h ahi = *(const v8h*)&At[arow][16 * l16 + 8];
    v16h af = CAT16(alo, ahi);
    v16h bf[4];
#pragma unroll
    for (int nt = 0; nt < 4; ++nt) {
      const int brow = nt * 16 + lm;
      v8h blo = *(const v8h*)&BtT[brow][16 * l16];
      v8h bhi = *(const v8h*)&BtT[brow][16 * l16 + 8];
      bf[nt] = CAT16(blo, bhi);
    }
#pragma unroll
    for (int nt = 0; nt < 4; ++nt) acc[nt] = WMMA_F16(af, bf[nt], acc[nt]);
  }

#pragma unroll
  for (int nt = 0; nt < 4; ++nt) {
    const int n = n0 + nt * 16 + lm;
    const float bias = bo[n];
#pragma unroll
    for (int r = 0; r < 8; ++r) {
      const int row = m0 + wid * 16 + r + 8 * l16;
      out[(size_t)row * kQD + n] = acc[nt][r] + bias;
    }
  }
}

// ---------------------------------------------------------------------------
extern "C" void kernel_launch(void* const* d_in, const int* in_sizes, int n_in,
                              void* d_out, int out_size, void* d_ws, size_t ws_size,
                              hipStream_t stream) {
  (void)in_sizes; (void)n_in; (void)out_size; (void)ws_size;
  const float* x  = (const float*)d_in[0];
  const float* Wq = (const float*)d_in[1];
  const float* Wk = (const float*)d_in[2];
  const float* Wv = (const float*)d_in[3];
  const float* Wo = (const float*)d_in[4];
  const float* bo = (const float*)d_in[5];
  float* out = (float*)d_out;

  char* ws = (char*)d_ws;
  size_t off = 0;
  auto carve = [&](size_t bytes) -> void* {
    void* p = ws + off;
    off = (off + bytes + 255) & ~(size_t)255;
    return p;
  };
  _Float16* xh   = (_Float16*)carve((size_t)kM  * kQD * 2);  // x in f16
  _Float16* wqh  = (_Float16*)carve((size_t)kQD * kIn * 2);
  _Float16* wkh  = (_Float16*)carve((size_t)kQD * kIn * 2);
  _Float16* wvh  = (_Float16*)carve((size_t)kQD * kIn * 2);
  _Float16* woh  = (_Float16*)carve((size_t)kIn * kQD * 2);
  _Float16* qws  = (_Float16*)carve((size_t)kM * kIn * 2);   // [b,h,s,64], pre-scaled
  _Float16* kws  = (_Float16*)carve((size_t)kM * kIn * 2);   // [b,h,s,64]
  _Float16* vtws = (_Float16*)carve((size_t)kM * kIn * 2);   // [b,h,64,s]
  _Float16* aws  = (_Float16*)carve((size_t)kM * kIn * 2);   // attn out [b*s, 512]

  const int nx = kM * kQD, nw = kQD * kIn;
  cvt_f32_f16<<<dim3((nx + 255) / 256), dim3(256), 0, stream>>>(x, xh, nx);
  cvt_f32_f16<<<dim3((nw + 255) / 256), dim3(256), 0, stream>>>(Wq, wqh, nw);
  cvt_f32_f16<<<dim3((nw + 255) / 256), dim3(256), 0, stream>>>(Wk, wkh, nw);
  cvt_f32_f16<<<dim3((nw + 255) / 256), dim3(256), 0, stream>>>(Wv, wvh, nw);
  cvt_f32_f16<<<dim3((nw + 255) / 256), dim3(256), 0, stream>>>(Wo, woh, nw);

  qkv_gemm<<<dim3(kM / 64, kIn / 64, 3), dim3(128), 0, stream>>>(
      xh, wqh, wkh, wvh, qws, kws, vtws);

  flash_attn<<<dim3(kS / 64, kB * kH), dim3(128), 0, stream>>>(qws, kws, vtws, aws);

  out_gemm<<<dim3(kM / 64, kQD / 64), dim3(128), 0, stream>>>(aws, woh, bo, out);
}
